// GAT_31044023615605
// MI455X (gfx1250) — matrix operand order repaired and checked by reading, not compiled, not run on hardware
//
#include <hip/hip_runtime.h>
#include <math.h>

typedef __bf16 bf16;
typedef __attribute__((ext_vector_type(16))) __bf16 v16bf;
typedef __attribute__((ext_vector_type(8)))  float  v8f;

#define N_NODES 4096
#define FIN     512
#define HID     64
#define HEADS   8
#define FOUT    256
#define LALPHA  0.2f
#define NEGV    (-9e15f)

static __device__ __forceinline__ float elu1(float x) {
    return x > 0.f ? x : expm1f(x);
}

// ---------------------------------------------------------------------------
// Pack adj (int32 0/1) into bitmask [N][128] words; wave-ballot, 1 word/wave.
// grid = (N, 16), block = 256 (8 waves -> 8 words).
// ---------------------------------------------------------------------------
__global__ __launch_bounds__(256)
void pack_adj_kernel(const int* __restrict__ adj, unsigned int* __restrict__ bits)
{
    const int row  = blockIdx.x;
    const int wid  = threadIdx.x >> 5;
    const int lane = threadIdx.x & 31;
    const int word = blockIdx.y * 8 + wid;          // 0..127
    const int col  = word * 32 + lane;
    unsigned long long b = __ballot(adj[(size_t)row * N_NODES + col] > 0);
    if (lane == 0) bits[(size_t)row * 128 + word] = (unsigned int)b;
}

// ---------------------------------------------------------------------------
// C[M,F] = A[M,K] * B[K,F]; fp32 in, bf16 WMMA, fp32 accumulate.
// Also writes CT[f][m] = bf16(C) (transposed bf16 copy for attention B-frags).
// block = 128 (4 waves, each owns a 16-row tile); grid = (M/64, F/64).
// ---------------------------------------------------------------------------
__global__ __launch_bounds__(128)
void gemm_bf16_kernel(const float* __restrict__ A, int lda,
                      const float* __restrict__ B, int ldb,
                      float* __restrict__ C, int ldc,
                      bf16* __restrict__ CT, int ldct,
                      int K)
{
    __shared__ float Bs[32][64];
    const int tid  = threadIdx.x;
    const int lane = tid & 31;
    const int wid  = tid >> 5;
    const int half = lane >> 4;
    const int l16  = lane & 15;
    const int rowT = blockIdx.x * 64 + wid * 16;
    const int colT = blockIdx.y * 64;

    v8f acc[4] = {};

    for (int k0 = 0; k0 < K; k0 += 32) {
        {
            const int r = tid >> 2;
            const int c = (tid & 3) * 16;
            const float4* src = reinterpret_cast<const float4*>(
                &B[(size_t)(k0 + r) * ldb + colT + c]);
            float4* dst = reinterpret_cast<float4*>(&Bs[r][c]);
            dst[0] = src[0]; dst[1] = src[1]; dst[2] = src[2]; dst[3] = src[3];
        }
        __syncthreads();

        // A fragment: half 0 -> K {0..7,16..23}; half 1 -> {8..15,24..31}
        v16bf a;
        {
            const float* arow = &A[(size_t)(rowT + l16) * lda + k0];
            const int kb1 = half * 8;
            const int kb2 = 16 + half * 8;
            float4 f0 = *reinterpret_cast<const float4*>(arow + kb1);
            float4 f1 = *reinterpret_cast<const float4*>(arow + kb1 + 4);
            float4 f2 = *reinterpret_cast<const float4*>(arow + kb2);
            float4 f3 = *reinterpret_cast<const float4*>(arow + kb2 + 4);
            a[0]=(bf16)f0.x; a[1]=(bf16)f0.y; a[2]=(bf16)f0.z; a[3]=(bf16)f0.w;
            a[4]=(bf16)f1.x; a[5]=(bf16)f1.y; a[6]=(bf16)f1.z; a[7]=(bf16)f1.w;
            a[8]=(bf16)f2.x; a[9]=(bf16)f2.y; a[10]=(bf16)f2.z; a[11]=(bf16)f2.w;
            a[12]=(bf16)f3.x; a[13]=(bf16)f3.y; a[14]=(bf16)f3.z; a[15]=(bf16)f3.w;
        }
        #pragma unroll
        for (int t = 0; t < 4; ++t) {
            v16bf b;
            #pragma unroll
            for (int i = 0; i < 16; ++i)
                b[i] = (bf16)Bs[(i & 8) * 2 + half * 8 + (i & 7)][t * 16 + l16];
            acc[t] = __builtin_amdgcn_wmma_f32_16x16x32_bf16(
                false, a, false, b, (short)0, acc[t], false, false);
        }
        __syncthreads();
    }

    #pragma unroll
    for (int t = 0; t < 4; ++t) {
        #pragma unroll
        for (int v = 0; v < 8; ++v) {
            const int r = rowT + v + 8 * half;
            const int c = colT + t * 16 + l16;
            const float val = acc[t][v];
            C[(size_t)r * ldc + c] = val;
            CT[(size_t)c * ldct + r] = (bf16)val;
        }
    }
}

// ---------------------------------------------------------------------------
// s1[row] = Wh[row]·a1, s2[row] = Wh[row]·a2. One wave per row.
// ---------------------------------------------------------------------------
__global__ __launch_bounds__(32)
void row_scores_kernel(const float* __restrict__ Wh, int ld, int F,
                       const float* __restrict__ a1, const float* __restrict__ a2,
                       float* __restrict__ s1, float* __restrict__ s2)
{
    const int row  = blockIdx.x;
    const int lane = threadIdx.x;
    const float* w = Wh + (size_t)row * ld;
    float p1 = 0.f, p2 = 0.f;
    for (int f = lane; f < F; f += 32) {
        const float v = w[f];
        p1 += v * a1[f];
        p2 += v * a2[f];
    }
    for (int o = 16; o > 0; o >>= 1) {
        p1 += __shfl_down(p1, o);
        p2 += __shfl_down(p2, o);
    }
    if (lane == 0) { s1[row] = p1; s2[row] = p2; }
}

// ---------------------------------------------------------------------------
// Per (row, head): online softmax stats over masked leaky-relu(s1_i + s2_j).
// grid = (N, H), block = 256.
// ---------------------------------------------------------------------------
__global__ __launch_bounds__(256)
void softmax_stats_kernel(const float* __restrict__ s1, const float* __restrict__ s2,
                          const unsigned int* __restrict__ bits,
                          float* __restrict__ rmax, float* __restrict__ rsum)
{
    __shared__ float sm[256];
    __shared__ float sl[256];
    const int i = blockIdx.x;
    const int h = blockIdx.y;
    const int t = threadIdx.x;
    const float s1v = s1[(size_t)h * N_NODES + i];
    const float* s2h = s2 + (size_t)h * N_NODES;
    const unsigned int* brow = bits + (size_t)i * 128;

    float m = NEGV, l = 0.f;
    for (int j = t; j < N_NODES; j += 256) {
        const float a = s1v + s2h[j];
        float e = a > 0.f ? a : LALPHA * a;
        if (!((brow[j >> 5] >> (j & 31)) & 1u)) e = NEGV;
        const float nm = fmaxf(m, e);
        l = l * __expf(m - nm) + __expf(e - nm);
        m = nm;
    }
    sm[t] = m; sl[t] = l;
    __syncthreads();
    for (int s = 128; s > 0; s >>= 1) {
        if (t < s) {
            const float m2 = sm[t + s], l2 = sl[t + s];
            const float nm = fmaxf(sm[t], m2);
            sl[t] = sl[t] * __expf(sm[t] - nm) + l2 * __expf(m2 - nm);
            sm[t] = nm;
        }
        __syncthreads();
    }
    if (t == 0) {
        rmax[(size_t)h * N_NODES + i] = sm[0];
        rsum[(size_t)h * N_NODES + i] = sl[0];
    }
}

// ---------------------------------------------------------------------------
// out[i][:] = elu( softmax_row(e) @ Wh ), attention probs built on the fly.
// NT = number of 16-wide feature tiles (covers full F per block).
// WhT j-tiles are double-buffer staged into LDS with async global->LDS DMA
// (ASYNCcnt-tracked), shared by all 4 waves of the block.
// block = 128 (4 waves x 16-row tiles); grid = (N/64, 1, heads).
// ---------------------------------------------------------------------------
template <int NT>
__global__ __launch_bounds__(128)
void attn_av_kernel(const float* __restrict__ s1, const float* __restrict__ s2,
                    const float* __restrict__ rmax, const float* __restrict__ rsum,
                    const unsigned int* __restrict__ bits,
                    const bf16* __restrict__ WhT,
                    size_t headStrideS, size_t headStrideW,
                    float* __restrict__ out, int ldOut, int headColStride)
{
    constexpr int NR  = NT * 16;        // rows (features) staged per j-tile
    constexpr int CPT = NR * 4 / 128;   // 16B async chunks per thread per tile
    // rows padded to 80B: word stride 20 -> conflict-free b128 reads, 16B aligned
    __shared__ __align__(16) bf16 Wbuf[2][NR][40];

    const int tid  = threadIdx.x;
    const int lane = tid & 31;
    const int wid  = tid >> 5;
    const int half = lane >> 4;
    const int l16  = lane & 15;
    const int hd   = blockIdx.z;
    const int rowT = blockIdx.x * 64 + wid * 16;

    const float* s1h = s1 + hd * headStrideS;
    const float* s2h = s2 + hd * headStrideS;
    const float* rmh = rmax + hd * headStrideS;
    const float* rsh = rsum + hd * headStrideS;
    const bf16*  WT  = WhT + hd * headStrideW;
    const int colBase = hd * headColStride;

    const int   i   = rowT + l16;            // A-frag row handled by this lane
    const float s1v = s1h[i];
    const float mx  = rmh[i];
    const unsigned int* brow = bits + (size_t)i * 128;

    const unsigned ldsOff = (unsigned)(size_t)(&Wbuf[0][0][0]);

    // stage WhT[0..NR-1][jj..jj+31] (bf16) into Wbuf[buf] via async DMA
    auto stage = [&](int buf, int jj) {
        const unsigned base = ldsOff + (unsigned)(buf * NR * 80);
        #pragma unroll
        for (int k = 0; k < CPT; ++k) {
            const int c   = tid + 128 * k;   // chunk id: 4 x 16B per row
            const int f   = c >> 2;
            const int seg = c & 3;
            const unsigned long long g = (unsigned long long)(size_t)
                (WT + (size_t)f * N_NODES + jj + seg * 8);
            const unsigned l = base + (unsigned)(f * 80 + seg * 16);
            asm volatile("global_load_async_to_lds_b128 %0, %1, off"
                         :: "v"(l), "v"(g) : "memory");
        }
    };

    v8f acc[NT] = {};

    stage(0, 0);
    for (int j0 = 0, it = 0; j0 < N_NODES; j0 += 32, ++it) {
        const int cur = it & 1;
        if (j0 + 32 < N_NODES) {
            stage(cur ^ 1, j0 + 32);        // overlap next tile's DMA
            if (NT == 16) asm volatile("s_wait_asynccnt 0x8" ::: "memory");
            else          asm volatile("s_wait_asynccnt 0x2" ::: "memory");
        } else {
            asm volatile("s_wait_asynccnt 0x0" ::: "memory");
        }
        __syncthreads();                    // tile `cur` visible to all waves

        const int jb1 = j0 + half * 8;
        const int jb2 = j0 + 16 + half * 8;
        if (j0 + 64 < N_NODES) __builtin_prefetch(s2h + j0 + 64, 0, 1);

        const unsigned int w = brow[j0 >> 5];
        const float4 q0 = *reinterpret_cast<const float4*>(s2h + jb1);
        const float4 q1 = *reinterpret_cast<const float4*>(s2h + jb1 + 4);
        const float4 q2 = *reinterpret_cast<const float4*>(s2h + jb2);
        const float4 q3 = *reinterpret_cast<const float4*>(s2h + jb2 + 4);

        auto pv = [&](unsigned int bit, float s2v) -> bf16 {
            const float a = s1v + s2v;
            float e = a > 0.f ? a : LALPHA * a;
            if (!bit) e = NEGV;
            return (bf16)__expf(e - mx);    // <= 1; masked underflows to 0
        };
        v16bf p;
        p[0]  = pv((w >> (half*8 + 0)) & 1u, q0.x);
        p[1]  = pv((w >> (half*8 + 1)) & 1u, q0.y);
        p[2]  = pv((w >> (half*8 + 2)) & 1u, q0.z);
        p[3]  = pv((w >> (half*8 + 3)) & 1u, q0.w);
        p[4]  = pv((w >> (half*8 + 4)) & 1u, q1.x);
        p[5]  = pv((w >> (half*8 + 5)) & 1u, q1.y);
        p[6]  = pv((w >> (half*8 + 6)) & 1u, q1.z);
        p[7]  = pv((w >> (half*8 + 7)) & 1u, q1.w);
        p[8]  = pv((w >> (16 + half*8 + 0)) & 1u, q2.x);
        p[9]  = pv((w >> (16 + half*8 + 1)) & 1u, q2.y);
        p[10] = pv((w >> (16 + half*8 + 2)) & 1u, q2.z);
        p[11] = pv((w >> (16 + half*8 + 3)) & 1u, q2.w);
        p[12] = pv((w >> (16 + half*8 + 4)) & 1u, q3.x);
        p[13] = pv((w >> (16 + half*8 + 5)) & 1u, q3.y);
        p[14] = pv((w >> (16 + half*8 + 6)) & 1u, q3.z);
        p[15] = pv((w >> (16 + half*8 + 7)) & 1u, q3.w);

        #pragma unroll
        for (int t = 0; t < NT; ++t) {
            // B-frag: lane holds feature col f, K contiguous in LDS row
            const bf16* brw = &Wbuf[cur][t * 16 + l16][half * 8];
            union { v16bf v; uint4 q[2]; } bb;
            bb.q[0] = *reinterpret_cast<const uint4*>(brw);
            bb.q[1] = *reinterpret_cast<const uint4*>(brw + 16);
            acc[t] = __builtin_amdgcn_wmma_f32_16x16x32_bf16(
                false, p, false, bb.v, (short)0, acc[t], false, false);
        }
        __syncthreads();                    // all reads done before buf reuse
    }

    #pragma unroll
    for (int t = 0; t < NT; ++t) {
        #pragma unroll
        for (int v = 0; v < 8; ++v) {
            const int r = rowT + v + 8 * half;
            float val = acc[t][v] / rsh[r];
            val = elu1(val);
            out[(size_t)r * ldOut + colBase + t * 16 + l16] = val;
        }
    }
}

// ---------------------------------------------------------------------------
// In-place row log-softmax over FOUT=256 features. grid = N, block = 256.
// ---------------------------------------------------------------------------
__global__ __launch_bounds__(256)
void log_softmax_kernel(float* __restrict__ out)
{
    __shared__ float red[256];
    const int i = blockIdx.x;
    const int t = threadIdx.x;
    const float x = out[(size_t)i * FOUT + t];
    red[t] = x;
    __syncthreads();
    for (int s = 128; s > 0; s >>= 1) {
        if (t < s) red[t] = fmaxf(red[t], red[t + s]);
        __syncthreads();
    }
    const float m = red[0];
    __syncthreads();
    red[t] = __expf(x - m);
    __syncthreads();
    for (int s = 128; s > 0; s >>= 1) {
        if (t < s) red[t] += red[t + s];
        __syncthreads();
    }
    const float lse = m + logf(red[0]);
    out[(size_t)i * FOUT + t] = x - lse;
}

// ---------------------------------------------------------------------------
extern "C" void kernel_launch(void* const* d_in, const int* in_sizes, int n_in,
                              void* d_out, int out_size, void* d_ws, size_t ws_size,
                              hipStream_t stream)
{
    (void)in_sizes; (void)n_in; (void)out_size; (void)ws_size;
    const float* x       = (const float*)d_in[0];
    const int*   adj     = (const int*)d_in[1];
    const float* W_heads = (const float*)d_in[2];
    const float* a1_h    = (const float*)d_in[3];
    const float* a2_h    = (const float*)d_in[4];
    const float* W_out   = (const float*)d_in[5];
    const float* a1_o    = (const float*)d_in[6];
    const float* a2_o    = (const float*)d_in[7];
    float* out = (float*)d_out;

    char* ws = (char*)d_ws;
    auto alloc = [&](size_t bytes) -> char* {
        char* p = ws;
        ws += (bytes + 255) & ~(size_t)255;
        return p;
    };

    unsigned int* bits = (unsigned int*)alloc((size_t)N_NODES * 128 * 4);        // 2 MB
    float* Wh   = (float*)alloc((size_t)HEADS * N_NODES * HID * 4);              // 8 MB
    bf16*  WhT  = (bf16*) alloc((size_t)HEADS * HID * N_NODES * 2);              // 4 MB
    float* s1_1 = (float*)alloc((size_t)HEADS * N_NODES * 4);
    float* s2_1 = (float*)alloc((size_t)HEADS * N_NODES * 4);
    float* rm1  = (float*)alloc((size_t)HEADS * N_NODES * 4);
    float* rs1  = (float*)alloc((size_t)HEADS * N_NODES * 4);
    float* hbuf = (float*)alloc((size_t)N_NODES * (HEADS * HID) * 4);            // 8 MB
    float* Wh2  = (float*)alloc((size_t)N_NODES * FOUT * 4);                     // 4 MB
    bf16*  Wh2T = (bf16*) alloc((size_t)FOUT * N_NODES * 2);                     // 2 MB
    float* s1_2 = (float*)alloc((size_t)N_NODES * 4);
    float* s2_2 = (float*)alloc((size_t)N_NODES * 4);
    float* rm2  = (float*)alloc((size_t)N_NODES * 4);
    float* rs2  = (float*)alloc((size_t)N_NODES * 4);

    // 0) pack adjacency into L2-resident bitmask
    pack_adj_kernel<<<dim3(N_NODES, 16), 256, 0, stream>>>(adj, bits);

    // 1) layer-1 per-head feature GEMMs + attention scores
    for (int hd = 0; hd < HEADS; ++hd) {
        gemm_bf16_kernel<<<dim3(N_NODES / 64, HID / 64), 128, 0, stream>>>(
            x, FIN,
            W_heads + (size_t)hd * FIN * HID, HID,
            Wh + (size_t)hd * N_NODES * HID, HID,
            WhT + (size_t)hd * HID * N_NODES, N_NODES,
            FIN);
        row_scores_kernel<<<N_NODES, 32, 0, stream>>>(
            Wh + (size_t)hd * N_NODES * HID, HID, HID,
            a1_h + hd * HID, a2_h + hd * HID,
            s1_1 + (size_t)hd * N_NODES, s2_1 + (size_t)hd * N_NODES);
    }
    softmax_stats_kernel<<<dim3(N_NODES, HEADS), 256, 0, stream>>>(
        s1_1, s2_1, bits, rm1, rs1);
    attn_av_kernel<4><<<dim3(N_NODES / 64, 1, HEADS), 128, 0, stream>>>(
        s1_1, s2_1, rm1, rs1, bits, WhT,
        (size_t)N_NODES, (size_t)HID * N_NODES,
        hbuf, HEADS * HID, HID);

    // 2) layer-2
    gemm_bf16_kernel<<<dim3(N_NODES / 64, FOUT / 64), 128, 0, stream>>>(
        hbuf, HEADS * HID, W_out, FOUT, Wh2, FOUT, Wh2T, N_NODES, HEADS * HID);
    row_scores_kernel<<<N_NODES, 32, 0, stream>>>(
        Wh2, FOUT, FOUT, a1_o, a2_o, s1_2, s2_2);
    softmax_stats_kernel<<<dim3(N_NODES, 1), 256, 0, stream>>>(
        s1_2, s2_2, bits, rm2, rs2);
    attn_av_kernel<16><<<dim3(N_NODES / 64, 1, 1), 128, 0, stream>>>(
        s1_2, s2_2, rm2, rs2, bits, Wh2T, 0, 0, out, FOUT, 0);

    // 3) final log-softmax (in place on d_out)
    log_softmax_kernel<<<N_NODES, 256, 0, stream>>>(out);
}
